// Attention_LoRA_48352741818398
// MI455X (gfx1250) — matrix operand order repaired and checked by reading, not compiled, not run on hardware
//
#include <hip/hip_runtime.h>
#include <hip/hip_bf16.h>

typedef __bf16 bf16_t;
typedef __attribute__((ext_vector_type(16))) __bf16 v16bf;
typedef __attribute__((ext_vector_type(8)))  __bf16 v8bf;
typedef __attribute__((ext_vector_type(8)))  float  v8f;
typedef __attribute__((ext_vector_type(4)))  int    v4i;

#define Bsz 2
#define Ssz 2048
#define Dsz 2048
#define Hn  16
#define HD  128
#define Rr  8
#define SCALING 2.0f   // 16.0 / R

// ---------------------------------------------------------------------------
// CDNA5 async global->LDS path (ASYNCcnt), with portable fallback
// ---------------------------------------------------------------------------
#if defined(__has_builtin)
#if __has_builtin(__builtin_amdgcn_global_load_async_to_lds_b128) && \
    __has_builtin(__builtin_amdgcn_s_wait_asynccnt)
#define USE_ASYNC 1
#endif
#endif

typedef __attribute__((address_space(1))) v4i as1_v4i;
typedef __attribute__((address_space(3))) v4i as3_v4i;

__device__ inline void async_copy16(const bf16_t* __restrict__ g, bf16_t* l) {
#ifdef USE_ASYNC
  __builtin_amdgcn_global_load_async_to_lds_b128(
      (as1_v4i*)(g), (as3_v4i*)(l), 0, 0);
#else
  *(v8bf*)l = *(const v8bf*)g;
#endif
}

template <int N>
__device__ inline void wait_async() {
#ifdef USE_ASYNC
  __builtin_amdgcn_s_wait_asynccnt(N);
#endif
}

// ---------------------------------------------------------------------------
// WMMA helpers (CDNA5 16x16x32 bf16, f32 accumulate)
// ---------------------------------------------------------------------------
__device__ inline v8f wmma_bf16(v16bf a, v16bf b, v8f c) {
  return __builtin_amdgcn_wmma_f32_16x16x32_bf16(false, a, false, b, (short)0, c,
                                                 false, false);
}

// A-matrix 16x32 bf16 fragment from row-major [*, ld] (global or LDS):
__device__ inline v16bf load_frag_A(const bf16_t* __restrict__ base, int ld,
                                    int row0, int k0) {
  int lane = threadIdx.x & 31;
  int m    = lane & 15;
  int h    = lane >> 4;
  const bf16_t* p = base + (size_t)(row0 + m) * ld + k0 + h * 8;
  v8bf lo = *(const v8bf*)p;
  v8bf hi = *(const v8bf*)(p + 16);
  v16bf f;
#pragma unroll
  for (int i = 0; i < 8; ++i) { f[i] = lo[i]; f[i + 8] = hi[i]; }
  return f;
}

// B-matrix 32x16 bf16 fragment; source row (col0+n) of K-contiguous [N,K]:
__device__ inline v16bf load_frag_B(const bf16_t* __restrict__ base, int ld,
                                    int col0, int k0) {
  int lane = threadIdx.x & 31;
  int n    = lane & 15;
  int h    = lane >> 4;
  const bf16_t* p = base + (size_t)(col0 + n) * ld + k0 + h * 8;
  v8bf lo = *(const v8bf*)p;
  v8bf hi = *(const v8bf*)(p + 16);
  v16bf f;
#pragma unroll
  for (int i = 0; i < 8; ++i) { f[i] = lo[i]; f[i + 8] = hi[i]; }
  return f;
}

// ---------------------------------------------------------------------------
// Kernel 1: LoRA-fold + transpose + bf16 cast, LDS-tiled (coalesced both sides)
// Wt[n,k] = bf16( W[k,n] + 2 * sum_r A[k,r]*B[r,n] )
// ---------------------------------------------------------------------------
__global__ __launch_bounds__(256) void merge_t_kernel(
    const float* __restrict__ W, const float* __restrict__ Am,
    const float* __restrict__ Bm, bf16_t* __restrict__ Wt) {
  __shared__ float tile[64][65];
  int tx = threadIdx.x & 63;
  int ty = threadIdx.x >> 6;        // 0..3
  int k0 = blockIdx.y * 64;         // W row block
  int n0 = blockIdx.x * 64;         // W col block
#pragma unroll
  for (int i = 0; i < 16; ++i) {
    int r = ty + i * 4;
    int k = k0 + r, n = n0 + tx;
    float v = W[(size_t)k * Dsz + n];
    float lora = 0.f;
#pragma unroll
    for (int rr = 0; rr < Rr; ++rr) lora += Am[k * Rr + rr] * Bm[rr * Dsz + n];
    tile[r][tx] = v + SCALING * lora;
  }
  __syncthreads();
#pragma unroll
  for (int i = 0; i < 16; ++i) {
    int c = ty + i * 4;
    Wt[(size_t)(n0 + c) * Dsz + k0 + tx] = (bf16_t)tile[tx][c];
  }
}

// ---------------------------------------------------------------------------
// Kernel 2: cast fp32 activations to bf16
// ---------------------------------------------------------------------------
__global__ void cast_kernel(const float* __restrict__ x, bf16_t* __restrict__ xb,
                            int n) {
  int i = blockIdx.x * blockDim.x + threadIdx.x;
  if (i < n) xb[i] = (bf16_t)x[i];
}

// ---------------------------------------------------------------------------
// Kernel 3: bf16 WMMA GEMM with double-buffered async LDS staging.
//   C[4096,2048] = A[4096,2048] * Wt^T
//   mode 0: store bf16 head-major [B,H,S,HD];  mode 2: store f32 [B,S,D]
// Block 256 thr = 8 waves; block tile 256M x 64N; wave tile 32M x 64N; K-step 32
// ---------------------------------------------------------------------------
__global__ __launch_bounds__(256) void gemm_kernel(
    const bf16_t* __restrict__ A, const bf16_t* __restrict__ Wt,
    void* __restrict__ out, int mode) {
  __shared__ __align__(16) bf16_t Abuf[2][256 * 32];   // 32 KiB
  __shared__ __align__(16) bf16_t Bbuf[2][64 * 32];    //  8 KiB

  int tid = threadIdx.x;
  int w   = tid >> 5;
  int m0  = blockIdx.y * 256;
  int n0  = blockIdx.x * 64;

  // per-tile async stage: A = 256x32 (4 chunks/thread), B = 64x32 (1 chunk/thread)
  auto stage = [&](int buf, int k0) {
#pragma unroll
    for (int i = 0; i < 4; ++i) {
      int c = tid + i * 256;
      int r = c >> 2, off = (c & 3) * 8;
      async_copy16(A + (size_t)(m0 + r) * Dsz + k0 + off, &Abuf[buf][r * 32 + off]);
    }
    {
      int r = tid >> 2, off = (tid & 3) * 8;
      async_copy16(Wt + (size_t)(n0 + r) * Dsz + k0 + off, &Bbuf[buf][r * 32 + off]);
    }
  };

  v8f acc[2][4] = {};
  stage(0, 0);
  for (int k0 = 0, it = 0; k0 < Dsz; k0 += 32, ++it) {
    int cur = it & 1, nxt = cur ^ 1;
    bool more = (k0 + 32) < Dsz;
    if (more) {
      stage(nxt, k0 + 32);
      wait_async<5>();              // 5 ops/thread in flight for next tile
    } else {
      wait_async<0>();
    }
    __syncthreads();                // current tile visible to all waves

    v16bf af0 = load_frag_A(&Abuf[cur][0], 32, w * 32, 0);
    v16bf af1 = load_frag_A(&Abuf[cur][0], 32, w * 32 + 16, 0);
    v16bf bfr[4];
#pragma unroll
    for (int t = 0; t < 4; ++t) bfr[t] = load_frag_B(&Bbuf[cur][0], 32, t * 16, 0);
    __syncthreads();                // reads done; cur may be overwritten next it

#pragma unroll
    for (int t = 0; t < 4; ++t) {
      acc[0][t] = wmma_bf16(af0, bfr[t], acc[0][t]);
      acc[1][t] = wmma_bf16(af1, bfr[t], acc[1][t]);
    }
  }

  int lane = tid & 31, nn = lane & 15, hi = lane >> 4;
#pragma unroll
  for (int s = 0; s < 2; ++s) {
    int row0 = m0 + w * 32 + s * 16;
#pragma unroll
    for (int t = 0; t < 4; ++t) {
#pragma unroll
      for (int m = 0; m < 8; ++m) {
        int row = row0 + m + hi * 8;
        int col = n0 + t * 16 + nn;
        float v = acc[s][t][m];
        if (mode == 2) {
          ((float*)out)[(size_t)row * Dsz + col] = v;
        } else {
          int bb = row >> 11, sq = row & 2047;
          int h = col >> 7, hd = col & 127;
          ((bf16_t*)out)[(((size_t)(bb * Hn + h) * Ssz + sq) * HD + hd)] = (bf16_t)v;
        }
      }
    }
  }
}

// ---------------------------------------------------------------------------
// Kernel 4: in-place RoPE on head-major bf16 [B,H,S,HD]
// ---------------------------------------------------------------------------
__global__ void rope_kernel(bf16_t* __restrict__ t, const float* __restrict__ cp,
                            const float* __restrict__ sp) {
  int idx  = blockIdx.x * blockDim.x + threadIdx.x;  // over B*H*S*(HD/2)
  int j    = idx & 63;
  int rest = idx >> 6;          // bh*S + s
  int s    = rest & 2047;
  float c  = cp[s * 64 + j];
  float sn = sp[s * 64 + j];
  size_t base = (size_t)idx * 2;
  float e = (float)t[base];
  float o = (float)t[base + 1];
  t[base]     = (bf16_t)(e * c - o * sn);
  t[base + 1] = (bf16_t)(e * sn + o * c);
}

// ---------------------------------------------------------------------------
// Kernel 5: LDS-tiled V transpose per head: [BH][S][HD] -> [BH][HD][S]
// ---------------------------------------------------------------------------
__global__ __launch_bounds__(256) void transpose_v_kernel(
    const bf16_t* __restrict__ V, bf16_t* __restrict__ Vt) {
  __shared__ bf16_t tile[64][65];
  int tx = threadIdx.x & 63, ty = threadIdx.x >> 6;
  int bh = blockIdx.z;
  int s0 = blockIdx.y * 64;
  int h0 = blockIdx.x * 64;
  const bf16_t* Vb = V + (size_t)bh * Ssz * HD;
  bf16_t* Vtb = Vt + (size_t)bh * HD * Ssz;
#pragma unroll
  for (int i = 0; i < 16; ++i) {
    int r = ty + i * 4;
    tile[r][tx] = Vb[(size_t)(s0 + r) * HD + h0 + tx];
  }
  __syncthreads();
#pragma unroll
  for (int i = 0; i < 16; ++i) {
    int c = ty + i * 4;
    Vtb[(size_t)(h0 + c) * Ssz + s0 + tx] = tile[tx][c];
  }
}

// ---------------------------------------------------------------------------
// Kernel 6: causal flash attention (WMMA). 1 wave = 16 q rows; 32 keys/step.
// Q,K: [B,H,S,HD] bf16; Vt: [B,H,HD,S] bf16; O: [B,S,H*HD] bf16
// ---------------------------------------------------------------------------
__global__ __launch_bounds__(256) void attn_kernel(
    const bf16_t* __restrict__ Q, const bf16_t* __restrict__ K,
    const bf16_t* __restrict__ Vt, bf16_t* __restrict__ O) {
  __shared__ __align__(64) bf16_t ptile[8][16 * 32];

  int w    = threadIdx.x >> 5;
  int lane = threadIdx.x & 31;
  int nn = lane & 15, hi = lane >> 4;
  int bh = blockIdx.x >> 4;       // b*H + h
  int qc = blockIdx.x & 15;
  int q0 = qc * 128 + w * 16;

  const bf16_t* Qh = Q  + (size_t)bh * Ssz * HD;
  const bf16_t* Kh = K  + (size_t)bh * Ssz * HD;
  const bf16_t* Vh = Vt + (size_t)bh * HD * Ssz;

  v16bf qf[4];
#pragma unroll
  for (int i = 0; i < 4; ++i) qf[i] = load_frag_A(Qh, HD, q0, 32 * i);

  v8f accO[8] = {};
  float Mrow[8], lrow[8];
#pragma unroll
  for (int m = 0; m < 8; ++m) { Mrow[m] = -1e30f; lrow[m] = 0.f; }

  const float cs = 0.08838834764831845f * 1.4426950408889634f; // 1/sqrt(128)*log2e

  for (int kv0 = 0; kv0 < q0 + 16; kv0 += 32) {
    if (kv0 + 32 < Ssz)
      __builtin_prefetch(Kh + (size_t)(kv0 + 32 + nn) * HD + hi * 64, 0, 1);

    v8f s0 = {}, s1 = {};
#pragma unroll
    for (int i = 0; i < 4; ++i) {
      v16bf kb0 = load_frag_B(Kh, HD, kv0,      32 * i);
      v16bf kb1 = load_frag_B(Kh, HD, kv0 + 16, 32 * i);
      s0 = wmma_bf16(qf[i], kb0, s0);
      s1 = wmma_bf16(qf[i], kb1, s1);
    }

    bool needmask = (kv0 + 31 > q0);
#pragma unroll
    for (int m = 0; m < 8; ++m) {
      int r = q0 + m + hi * 8;
      float a0 = s0[m] * cs;
      float a1 = s1[m] * cs;
      if (needmask) {
        if (kv0 + nn > r)      a0 = -1e30f;
        if (kv0 + 16 + nn > r) a1 = -1e30f;
      }
      float t = fmaxf(a0, a1);
      t = fmaxf(t, __shfl_xor(t, 1));
      t = fmaxf(t, __shfl_xor(t, 2));
      t = fmaxf(t, __shfl_xor(t, 4));
      t = fmaxf(t, __shfl_xor(t, 8));
      float mn = fmaxf(Mrow[m], t);
      float p0 = exp2f(a0 - mn);
      float p1 = exp2f(a1 - mn);
      float rs = p0 + p1;
      rs += __shfl_xor(rs, 1);
      rs += __shfl_xor(rs, 2);
      rs += __shfl_xor(rs, 4);
      rs += __shfl_xor(rs, 8);
      float corr = exp2f(Mrow[m] - mn);
      lrow[m] = lrow[m] * corr + rs;
      Mrow[m] = mn;
#pragma unroll
      for (int j = 0; j < 8; ++j) accO[j][m] *= corr;
      int rowl = m + hi * 8;
      ptile[w][rowl * 32 + nn]      = (bf16_t)p0;
      ptile[w][rowl * 32 + 16 + nn] = (bf16_t)p1;
    }

    const bf16_t* pp = &ptile[w][(lane & 15) * 32 + hi * 8];
    v8bf plo = *(const v8bf*)pp;
    v8bf phi = *(const v8bf*)(pp + 16);
    v16bf pa;
#pragma unroll
    for (int i = 0; i < 8; ++i) { pa[i] = plo[i]; pa[i + 8] = phi[i]; }

#pragma unroll
    for (int j = 0; j < 8; ++j) {
      v16bf vb = load_frag_B(Vh, Ssz, j * 16, kv0);
      accO[j] = wmma_bf16(pa, vb, accO[j]);
    }
  }

  int b = bh >> 4, h = bh & 15;
#pragma unroll
  for (int j = 0; j < 8; ++j) {
#pragma unroll
    for (int m = 0; m < 8; ++m) {
      int r = q0 + m + hi * 8;
      float v = accO[j][m] / lrow[m];
      O[((size_t)(b * Ssz + r)) * Dsz + h * HD + j * 16 + nn] = (bf16_t)v;
    }
  }
}

// ---------------------------------------------------------------------------
// Host-side orchestration
// ---------------------------------------------------------------------------
extern "C" void kernel_launch(void* const* d_in, const int* in_sizes, int n_in,
                              void* d_out, int out_size, void* d_ws, size_t ws_size,
                              hipStream_t stream) {
  const float* x    = (const float*)d_in[0];
  const float* wq   = (const float*)d_in[1];
  const float* wk   = (const float*)d_in[2];
  const float* wv   = (const float*)d_in[3];
  const float* wo   = (const float*)d_in[4];
  const float* wq_A = (const float*)d_in[5];
  const float* wq_B = (const float*)d_in[6];
  const float* wk_A = (const float*)d_in[7];
  const float* wk_B = (const float*)d_in[8];
  const float* wv_A = (const float*)d_in[9];
  const float* wv_B = (const float*)d_in[10];
  const float* wo_A = (const float*)d_in[11];
  const float* wo_B = (const float*)d_in[12];
  const float* cosp = (const float*)d_in[13];
  const float* sinp = (const float*)d_in[14];

  const size_t WSZ = (size_t)Dsz * Dsz * sizeof(bf16_t);          // 8 MiB
  const size_t ASZ = (size_t)Bsz * Ssz * Dsz * sizeof(bf16_t);    // 16 MiB
  char* ws = (char*)d_ws;
  bf16_t* wqt  = (bf16_t*)(ws);
  bf16_t* wkt  = (bf16_t*)(ws + WSZ);
  bf16_t* wvt  = (bf16_t*)(ws + 2 * WSZ);
  bf16_t* wot  = (bf16_t*)(ws + 3 * WSZ);
  bf16_t* xb   = (bf16_t*)(ws + 4 * WSZ);            // also reused as attn buf
  bf16_t* qbuf = (bf16_t*)(ws + 4 * WSZ + ASZ);
  bf16_t* kbuf = (bf16_t*)(ws + 4 * WSZ + 2 * ASZ);
  bf16_t* vbuf = (bf16_t*)(ws + 4 * WSZ + 3 * ASZ);
  bf16_t* vtb  = (bf16_t*)(ws + 4 * WSZ + 4 * ASZ);
  bf16_t* attn = xb;   // x no longer needed after the V projection

  const int nX = Bsz * Ssz * Dsz;                  // 8,388,608
  dim3 merge_grid(Dsz / 64, Dsz / 64);             // 32 x 32
  dim3 gemm_grid(Dsz / 64, (Bsz * Ssz) / 256);     // 32 x 16
  dim3 vt_grid(HD / 64, Ssz / 64, Bsz * Hn);       // 2 x 32 x 32

  // 1. fold LoRA into weights, transpose (LDS-tiled), cast bf16
  merge_t_kernel<<<merge_grid, 256, 0, stream>>>(wq, wq_A, wq_B, wqt);
  merge_t_kernel<<<merge_grid, 256, 0, stream>>>(wk, wk_A, wk_B, wkt);
  merge_t_kernel<<<merge_grid, 256, 0, stream>>>(wv, wv_A, wv_B, wvt);
  merge_t_kernel<<<merge_grid, 256, 0, stream>>>(wo, wo_A, wo_B, wot);

  // 2. cast activations
  cast_kernel<<<nX / 256, 256, 0, stream>>>(x, xb, nX);

  // 3. QKV projections (async-LDS WMMA GEMM), all head-major
  gemm_kernel<<<gemm_grid, 256, 0, stream>>>(xb, wqt, (void*)qbuf, 0);
  gemm_kernel<<<gemm_grid, 256, 0, stream>>>(xb, wkt, (void*)kbuf, 0);
  gemm_kernel<<<gemm_grid, 256, 0, stream>>>(xb, wvt, (void*)vbuf, 0);

  // 4. RoPE in place on Q and K; transpose V per head (LDS-tiled)
  const int nR = Bsz * Hn * Ssz * (HD / 2);        // 4,194,304
  rope_kernel<<<nR / 256, 256, 0, stream>>>(qbuf, cosp, sinp);
  rope_kernel<<<nR / 256, 256, 0, stream>>>(kbuf, cosp, sinp);
  transpose_v_kernel<<<vt_grid, 256, 0, stream>>>(vbuf, vtb);

  // 5. causal flash attention (WMMA)
  attn_kernel<<<Bsz * Hn * (Ssz / 128), 256, 0, stream>>>(qbuf, kbuf, vtb, attn);

  // 6. output projection with merged Wo' -> f32 d_out
  gemm_kernel<<<gemm_grid, 256, 0, stream>>>(attn, wot, d_out, 2);
}
